// Attention_66718021976771
// MI455X (gfx1250) — matrix operand order repaired and checked by reading, not compiled, not run on hardware
//
#include <hip/hip_runtime.h>
#include <hip/hip_bf16.h>

typedef __attribute__((ext_vector_type(16))) __bf16 v16bf;
typedef __attribute__((ext_vector_type(8)))  float  v8f;

#define Hdim 128
#define Bsz  32
#define Tlen 8192
#define TBL_ELEMS (5*8*32*16)   // [c=5][ntile=8][lane=32][e=16] bf16 fragments

#define BLOCKS 256
#define WAVES_PER_BLOCK 8
#define TOTAL_WAVES (BLOCKS * WAVES_PER_BLOCK)   // 2048
#define ROW_TILES ((Tlen * Bsz) / 32)            // 8192 tiles of 32 rows
#define ITERS (ROW_TILES / TOTAL_WAVES)          // 4 tiles per wave

// ---------------------------------------------------------------------------
// Kernel 0: hb[b][n] = h[b] . W_attn[n][0:128] + b_attn[n]   (tiny)
// ---------------------------------------------------------------------------
__global__ void k_hb(const float* __restrict__ hidden, const float* __restrict__ W,
                     const float* __restrict__ bias, float* __restrict__ hb) {
  int b = blockIdx.x;      // 32 blocks
  int n = threadIdx.x;     // 128 threads
  const float* h = hidden + b * Hdim;
  const float* w = W + n * (2 * Hdim);
  float s = bias[n];
#pragma unroll 8
  for (int k = 0; k < Hdim; ++k) s += h[k] * w[k];
  hb[b * Hdim + n] = s;
}

// ---------------------------------------------------------------------------
// Kernel 1: pack augmented B matrix [W2.T ; hb] (K=160, N=128) into bf16
// WMMA B-fragment layout: table[c][nt][lane][e], e=2g+h -> VGPR g lo/hi,
//   lane<16:  K = c*32 + e (e<8) / c*32 + 8 + e (e>=8),  N = nt*16 + lane
//   lane>=16: K = c*32 + 8 + e (e<8) / c*32 + 16 + e (e>=8), N = nt*16 + (lane-16)
// c==4 is the one-hot/bias chunk: B[128+j][N] = hb[j][N]
// ---------------------------------------------------------------------------
__global__ void k_pack(const float* __restrict__ W, const float* __restrict__ hb,
                       __bf16* __restrict__ table) {
  int idx = blockIdx.x * 256 + threadIdx.x;
  if (idx >= TBL_ELEMS) return;
  int e     = idx & 15;
  int lane  = (idx >> 4) & 31;
  int nt    = (idx >> 9) & 7;
  int c     = idx >> 12;
  int N     = nt * 16 + (lane & 15);
  int khalf = (lane >> 4) * 8 + ((e >= 8) ? 8 : 0);
  float val;
  if (c < 4) val = W[N * (2 * Hdim) + Hdim + c * 32 + khalf + e];  // W2[N][k] = B[k][N]
  else       val = hb[(khalf + e) * Hdim + N];
  table[idx] = (__bf16)val;
}

// ---------------------------------------------------------------------------
// Kernel 2: persistent GEMM + relu + dot(v).
// One wave = 32 flat rows per iteration (rows r = t*B+b are contiguous in
// memory; a 32-row aligned tile is exactly one t over all 32 batch rows).
// Two 16-row A-fragment sets share each B fragment -> 2 wmma per ds pair,
// two independent accumulation chains.  K=160 = 128 data + 32 one-hot bias.
// ---------------------------------------------------------------------------
__global__ void __launch_bounds__(256) k_main(const float* __restrict__ enc,
                                              const __bf16* __restrict__ table,
                                              const float* __restrict__ v,
                                              float* __restrict__ S) {
  __shared__ v16bf ldsB[5 * 8 * 32];   // 40 KB fragment table, staged once
  __shared__ float ldsv[Hdim];

  {
    const uint4* src = (const uint4*)table;
    uint4* dst = (uint4*)ldsB;
    for (int i = threadIdx.x; i < (TBL_ELEMS * 2) / 16; i += 256) dst[i] = src[i];
    if (threadIdx.x < Hdim) ldsv[threadIdx.x] = v[threadIdx.x];
  }
  __syncthreads();

  const int lane  = threadIdx.x & 31;
  const int wave  = threadIdx.x >> 5;
  const int gwave = blockIdx.x * WAVES_PER_BLOCK + wave;
  const int koff  = (lane >> 4) * 8;   // 0 or 8: lane's K sub-block
  const int myrow = lane & 15;

  // Loop-invariant one-hot bias-select fragments (tile rows are 32-aligned,
  // so b == myrow for the low 16 rows and b == myrow+16 for the high 16).
  v16bf oh0, oh1;
#pragma unroll
  for (int e = 0; e < 16; ++e) {
    int j = koff + e + ((e >= 8) ? 8 : 0);
    oh0[e] = (j == myrow)      ? (__bf16)1.0f : (__bf16)0.0f;
    oh1[e] = (j == myrow + 16) ? (__bf16)1.0f : (__bf16)0.0f;
  }

  for (int it = 0; it < ITERS; ++it) {
    const long row_base = ((long)gwave + (long)it * TOTAL_WAVES) * 32;

    // ---- A fragments: 32 rows x 128 K, f32 -> bf16 in-register ----
    const float* rp0 = enc + (row_base + myrow) * Hdim;
    const float* rp1 = rp0 + 16 * Hdim;
    v16bf a0[5], a1[5];
#pragma unroll
    for (int c = 0; c < 4; ++c) {
      const float4* q0 = (const float4*)(rp0 + c * 32 + koff);
      const float4* q1 = (const float4*)(rp0 + c * 32 + koff + 16);
      float4 f0 = q0[0], f1 = q0[1], f2 = q1[0], f3 = q1[1];
      v16bf t;
      t[0]=(__bf16)f0.x;  t[1]=(__bf16)f0.y;  t[2]=(__bf16)f0.z;  t[3]=(__bf16)f0.w;
      t[4]=(__bf16)f1.x;  t[5]=(__bf16)f1.y;  t[6]=(__bf16)f1.z;  t[7]=(__bf16)f1.w;
      t[8]=(__bf16)f2.x;  t[9]=(__bf16)f2.y;  t[10]=(__bf16)f2.z; t[11]=(__bf16)f2.w;
      t[12]=(__bf16)f3.x; t[13]=(__bf16)f3.y; t[14]=(__bf16)f3.z; t[15]=(__bf16)f3.w;
      a0[c] = t;
      const float4* r0 = (const float4*)(rp1 + c * 32 + koff);
      const float4* r1 = (const float4*)(rp1 + c * 32 + koff + 16);
      float4 g0 = r0[0], g1 = r0[1], g2 = r1[0], g3 = r1[1];
      v16bf u;
      u[0]=(__bf16)g0.x;  u[1]=(__bf16)g0.y;  u[2]=(__bf16)g0.z;  u[3]=(__bf16)g0.w;
      u[4]=(__bf16)g1.x;  u[5]=(__bf16)g1.y;  u[6]=(__bf16)g1.z;  u[7]=(__bf16)g1.w;
      u[8]=(__bf16)g2.x;  u[9]=(__bf16)g2.y;  u[10]=(__bf16)g2.z; u[11]=(__bf16)g2.w;
      u[12]=(__bf16)g3.x; u[13]=(__bf16)g3.y; u[14]=(__bf16)g3.z; u[15]=(__bf16)g3.w;
      a1[c] = u;
    }
    a0[4] = oh0;
    a1[4] = oh1;

    float p0[8], p1[8];
#pragma unroll
    for (int g = 0; g < 8; ++g) { p0[g] = 0.f; p1[g] = 0.f; }

    // ---- 8 N-tiles x 5 K-chunks; each B fragment feeds 2 WMMAs ----
#pragma unroll 2
    for (int nt = 0; nt < 8; ++nt) {
      v8f acc0 = {}, acc1 = {};
#pragma unroll
      for (int c = 0; c < 5; ++c) {
        v16bf bf = ldsB[c * 256 + nt * 32 + lane];
        acc0 = __builtin_amdgcn_wmma_f32_16x16x32_bf16(
            false, a0[c], false, bf, (short)0, acc0, false, false);
        acc1 = __builtin_amdgcn_wmma_f32_16x16x32_bf16(
            false, a1[c], false, bf, (short)0, acc1, false, false);
      }
      float vc = ldsv[nt * 16 + myrow];
#pragma unroll
      for (int g = 0; g < 8; ++g) {
        float e0 = acc0[g]; e0 = e0 > 0.f ? e0 : 0.f; p0[g] += e0 * vc;
        float e1 = acc1[g]; e1 = e1 > 0.f ? e1 : 0.f; p1[g] += e1 * vc;
      }
    }

    // ---- reduce over 16 lanes (cols) within each half-wave ----
#pragma unroll
    for (int mask = 1; mask < 16; mask <<= 1)
#pragma unroll
      for (int g = 0; g < 8; ++g) {
        p0[g] += __shfl_xor(p0[g], mask, 32);
        p1[g] += __shfl_xor(p1[g], mask, 32);
      }

    // lane 0 holds rows +0..7 / +16..23, lane 16 holds +8..15 / +24..31;
    // tile == one t over all b:  b = r & 31,  t = row_base >> 5
    if (myrow == 0) {
      const int  half = (lane >> 4) * 8;
      const long t    = row_base >> 5;
#pragma unroll
      for (int g = 0; g < 8; ++g) {
        S[(long)(half + g)      * Tlen + t] = p0[g];
        S[(long)(half + g + 16) * Tlen + t] = p1[g];
      }
    }
  }
}

// ---------------------------------------------------------------------------
// Kernel 3: ragged masked softmax per batch row (B=32 blocks over T=8192)
// ---------------------------------------------------------------------------
__global__ void __launch_bounds__(256) k_softmax(const float* __restrict__ S,
                                                 const int* __restrict__ len_seq,
                                                 float* __restrict__ out) {
  int b = blockIdx.x;
  int L = len_seq[b];
  const float* s = S + (long)b * Tlen;
  float* o = out + (long)b * Tlen;
  __shared__ float red[256];
  int tid = threadIdx.x;

  float m = -3.4e38f;
  for (int t = tid; t < L; t += 256) m = fmaxf(m, s[t]);
  red[tid] = m; __syncthreads();
  for (int st = 128; st > 0; st >>= 1) {
    if (tid < st) red[tid] = fmaxf(red[tid], red[tid + st]);
    __syncthreads();
  }
  m = red[0];
  __syncthreads();

  float sum = 0.f;
  for (int t = tid; t < L; t += 256) sum += __expf(s[t] - m);
  red[tid] = sum; __syncthreads();
  for (int st = 128; st > 0; st >>= 1) {
    if (tid < st) red[tid] += red[tid + st];
    __syncthreads();
  }
  float inv = 1.0f / red[0];

  for (int t = tid; t < Tlen; t += 256)
    o[t] = (t < L) ? __expf(s[t] - m) * inv : 0.0f;
}

// ---------------------------------------------------------------------------
extern "C" void kernel_launch(void* const* d_in, const int* in_sizes, int n_in,
                              void* d_out, int out_size, void* d_ws, size_t ws_size,
                              hipStream_t stream) {
  (void)in_sizes; (void)n_in; (void)out_size; (void)ws_size;
  const float* hidden  = (const float*)d_in[0];   // [B,H]
  const float* enc     = (const float*)d_in[1];   // [T,B,H]
  const int*   len_seq = (const int*)  d_in[2];   // [B]
  const float* W       = (const float*)d_in[3];   // [H,2H]
  const float* bias    = (const float*)d_in[4];   // [H]
  const float* v       = (const float*)d_in[5];   // [H]
  float* out = (float*)d_out;                     // [B,1,T]

  char* ws = (char*)d_ws;
  float*  hb    = (float*)ws;                         // 32*128 f32   = 16 KB
  __bf16* table = (__bf16*)(ws + 16384);              // 20480 bf16   = 40 KB
  float*  S     = (float*)(ws + 16384 + 40960);       // 32*8192 f32  = 1 MB

  k_hb     <<<Bsz, Hdim, 0, stream>>>(hidden, W, bias, hb);
  k_pack   <<<(TBL_ELEMS + 255) / 256, 256, 0, stream>>>(W, hb, table);
  k_main   <<<BLOCKS, 256, 0, stream>>>(enc, table, v, S);
  k_softmax<<<Bsz, 256, 0, stream>>>(S, len_seq, out);
}
